// NonLocalChannelBlock2D_72155450572818
// MI455X (gfx1250) — matrix operand order repaired and checked by reading, not compiled, not run on hardware
//
#include <hip/hip_runtime.h>
#include <hip/hip_bf16.h>
#include <stdint.h>

typedef __attribute__((ext_vector_type(16))) __bf16       v16bf;
typedef __attribute__((ext_vector_type(8)))  float        v8f;
typedef __attribute__((ext_vector_type(8)))  unsigned int v8u;

#define DEVINL __device__ __forceinline__

// round-to-nearest-even f32 -> bf16 (bits)
DEVINL unsigned short f2bf(float f) {
  unsigned u = __builtin_bit_cast(unsigned, f);
  unsigned r = u + 0x7FFFu + ((u >> 16) & 1u);
  return (unsigned short)(r >> 16);
}

DEVINL v8f wmma_bf16(v8u a, v8u b, v8f c) {
  return __builtin_amdgcn_wmma_f32_16x16x32_bf16(
      false, __builtin_bit_cast(v16bf, a),
      false, __builtin_bit_cast(v16bf, b),
      (short)0, c, false, false);
}

// Per-lane K element offsets inside a 32-wide K chunk (bf16 = u32 pairs).
// A (16x32, MxK): lanes 0-15 hold M=lane; V0..3 K=hl*8+2v+h, V4..7 K=16+...
DEVINL void a_koffs(int hl, int* ko) {
#pragma unroll
  for (int v = 0; v < 8; ++v) ko[v] = ((v < 4) ? 0 : 16) + hl * 8 + 2 * (v & 3);
}
// B (32x16, KxN): lanes 0-15 hold N=lane; V0..3 K=hl*8+2v+h, V4..7 K=16+...
DEVINL void b_koffs(int hl, int* ko) {
#pragma unroll
  for (int v = 0; v < 8; ++v) ko[v] = ((v < 4) ? 0 : 16) + hl * 8 + 2 * (v & 3);
}

// ---------------------------------------------------------------------------
// Kernel 1: 2x2 maxpool [8,128,256,256] -> bf16 [8,128,128,128]
// ---------------------------------------------------------------------------
__global__ void __launch_bounds__(256)
k_pool(const float* __restrict__ x, unsigned short* __restrict__ xp) {
  int i = blockIdx.x * 256 + threadIdx.x;        // ((b*128+c)*128+oy)*128+ox
  int ox = i & 127;
  int t  = i >> 7;
  int oy = t & 127;
  t >>= 7;                                       // t = b*128 + c
  const float* p = x + (((size_t)t) << 16) + (size_t)(2 * oy) * 256 + 2 * ox;
  float m = fmaxf(fmaxf(p[0], p[1]), fmaxf(p[256], p[257]));
  xp[i] = f2bf(m);
}

// ---------------------------------------------------------------------------
// Kernel 1b: repack both conv weights f32 [O][I][3][3] -> bf16 [tap][O][I]
// (theta at offset 0, phi at offset 9*128*128). One-time; 590 KB output.
// ---------------------------------------------------------------------------
__global__ void __launch_bounds__(256)
k_repack(const float* __restrict__ thw, const float* __restrict__ phw,
         unsigned short* __restrict__ wr) {
  int i = blockIdx.x * 256 + threadIdx.x;        // 2 * 9*128*128
  int e = i;
  const float* src = thw;
  if (e >= 9 * 128 * 128) { e -= 9 * 128 * 128; src = phw; }
  int ci = e & 127, o = (e >> 7) & 127, tap = e >> 14;
  wr[i] = f2bf(src[(o * 128 + ci) * 9 + tap]);
}

// ---------------------------------------------------------------------------
// Kernels 2/3: 3x3 conv, SAME pad, implicit GEMM with bf16 WMMA.
// Block = 256 thr (8 waves): (batch, 16-out-ch tile, 2 output rows).
// Wave w: output row = ypair*2 + (w>>2), pixels (w&3)*32 .. +31 (2 accums).
// A fragments come straight from the L2-resident repacked weights; only the
// haloed input rows (4 x 130 x 32ch per K-chunk) are staged in LDS.
// ---------------------------------------------------------------------------
__global__ void __launch_bounds__(256)
k_conv3x3(const unsigned short* __restrict__ xp,
          const unsigned short* __restrict__ wrep,      // [tap][o][ci] bf16
          const float* __restrict__ bias, unsigned short* __restrict__ outb) {
  __shared__ unsigned short ilds[4 * 130 * 32];   // [row][col][ci32] bf16

  const int tid = threadIdx.x;
  const int bid = blockIdx.x;                    // 4096 = 8b * 8cht * 64 ypair
  const int yp  = bid & 63;
  const int cht = (bid >> 6) & 7;
  const int b   = bid >> 9;

  const int wave = tid >> 5, lane = tid & 31;
  const int nn = lane & 15, hl = lane >> 4;
  const int r0 = wave >> 2;                      // 0/1: output row in pair
  const int y  = yp * 2 + r0;
  const int px = (wave & 3) * 32;
  int kA[8], kB[8];
  a_koffs(hl, kA);
  b_koffs(hl, kB);
  v8f acc0 = {}, acc1 = {};

  for (int ch = 0; ch < 4; ++ch) {               // 4 chunks of 32 in-ch
    // stage 4 haloed rows (yp*2-1 .. yp*2+2), 130 cols, 32 channels
    for (int i = tid; i < 4 * 130 * 32; i += 256) {
      int c = i & 31, t2 = i >> 5;
      int col = t2 % 130, r = t2 / 130;
      int iy = yp * 2 + r - 1, ix = col - 1;
      unsigned short vv = 0;
      if ((unsigned)iy < 128u && (unsigned)ix < 128u)
        vv = xp[(((size_t)(b * 128 + ch * 32 + c)) << 14) + iy * 128 + ix];
      ilds[(r * 130 + col) * 32 + c] = vv;
    }
    __syncthreads();
    // prefetch next chunk's rows into L2 while we compute
    if (ch < 3) {
      int iy = yp * 2 + ((tid >> 5) & 3) - 1;
      if ((unsigned)iy < 128u)
        __builtin_prefetch(
            xp + (((size_t)(b * 128 + (ch + 1) * 32 + (tid & 31))) << 14) +
                iy * 128, 0, 0);
    }
#pragma unroll
    for (int tap = 0; tap < 9; ++tap) {
      const int ky = tap / 3, kx = tap % 3;
      const unsigned short* wa =
          wrep + ((size_t)(tap * 128 + cht * 16 + nn)) * 128 + ch * 32;
      const unsigned short* ib0 =
          &ilds[((r0 + ky) * 130 + px + nn + kx) * 32];
      v8u A, B0, B1;
#pragma unroll
      for (int v = 0; v < 8; ++v) A[v] = *(const unsigned int*)(wa + kA[v]);
#pragma unroll
      for (int v = 0; v < 8; ++v)
        B0[v] = *(const unsigned int*)(ib0 + kB[v]);
#pragma unroll
      for (int v = 0; v < 8; ++v)
        B1[v] = *(const unsigned int*)(ib0 + 16 * 32 + kB[v]);
      acc0 = wmma_bf16(A, B0, acc0);
      acc1 = wmma_bf16(A, B1, acc1);
    }
    __syncthreads();
  }
#pragma unroll
  for (int r = 0; r < 8; ++r) {
    int m = r + hl * 8;
    int c = cht * 16 + m;
    float bv = bias[c];
    size_t base = (((size_t)(b * 128 + c)) << 14) + y * 128;
    outb[base + px + nn]      = f2bf(acc0[r] + bv);
    outb[base + px + 16 + nn] = f2bf(acc1[r] + bv);
  }
}

// ---------------------------------------------------------------------------
// Kernel 4: f[b] = theta[b] (128x16384) @ phi[b]^T -> [128,128] f32.
// One wave per 16x16 tile, K = 16384, operands stream from L2-resident bf16.
// ---------------------------------------------------------------------------
__global__ void __launch_bounds__(256)
k_fmm(const unsigned short* __restrict__ th, const unsigned short* __restrict__ ph,
      float* __restrict__ f) {
  const int tid = threadIdx.x;
  const int wave = tid >> 5, lane = tid & 31;
  const int t = blockIdx.x * 8 + wave;           // 512 waves = 8b * 8i * 8j
  const int b = t >> 6, rem = t & 63, i = rem >> 3, j = rem & 7;
  const int nn = lane & 15, hl = lane >> 4;
  int kA[8], kB[8];
  a_koffs(hl, kA);
  b_koffs(hl, kB);
  const unsigned short* ta = th + (((size_t)(b * 128 + i * 16 + nn)) << 14);
  const unsigned short* pb = ph + (((size_t)(b * 128 + j * 16 + nn)) << 14);
  v8f acc = {};
  for (int k = 0; k < 16384; k += 32) {
    v8u A, B;
#pragma unroll
    for (int v = 0; v < 8; ++v) A[v] = *(const unsigned int*)(ta + k + kA[v]);
#pragma unroll
    for (int v = 0; v < 8; ++v) B[v] = *(const unsigned int*)(pb + k + kB[v]);
    acc = wmma_bf16(A, B, acc);
  }
#pragma unroll
  for (int r = 0; r < 8; ++r) {
    int m = r + hl * 8;
    f[((size_t)(b * 128 + i * 16 + m)) * 128 + j * 16 + nn] = acc[r];
  }
}

// ---------------------------------------------------------------------------
// Kernel 5: row softmax over f [1024 rows x 128], in place.
// ---------------------------------------------------------------------------
__global__ void __launch_bounds__(128) k_softmax(float* __restrict__ f) {
  __shared__ float red[128];
  const int row = blockIdx.x, tid = threadIdx.x;
  float v = f[(size_t)row * 128 + tid];
  red[tid] = v;
  __syncthreads();
  for (int s = 64; s > 0; s >>= 1) {
    if (tid < s) red[tid] = fmaxf(red[tid], red[tid + s]);
    __syncthreads();
  }
  float mx = red[0];
  __syncthreads();
  float e = __expf(v - mx);
  red[tid] = e;
  __syncthreads();
  for (int s = 64; s > 0; s >>= 1) {
    if (tid < s) red[tid] += red[tid + s];
    __syncthreads();
  }
  f[(size_t)row * 128 + tid] = e / red[0];
}

// ---------------------------------------------------------------------------
// Kernel 6: M[b] = W (128x128) @ a[b] (128x128), stored bf16. Tiny.
// ---------------------------------------------------------------------------
__global__ void __launch_bounds__(256)
k_wm(const float* __restrict__ Ww, const float* __restrict__ a,
     unsigned short* __restrict__ Mb) {
  int gid = blockIdx.x * 256 + threadIdx.x;      // ((b*128+o)*128+d)
  int d = gid & 127, o = (gid >> 7) & 127, b = gid >> 14;
  const float* wr = Ww + o * 128;
  const float* ac = a + ((size_t)b * 128) * 128 + d;
  float s = 0.f;
  for (int c = 0; c < 128; ++c) s += wr[c] * ac[(size_t)c * 128];
  Mb[gid] = f2bf(s);
}

// ---------------------------------------------------------------------------
// Kernel 7: out = (M[b] @ x_flat[b]) * bn_scale + (Wb*scale + shift) + x
// Block: 8 waves, 16 out-ch x 128 n; x tile (128k x 128n) bf16-packed in LDS.
// ---------------------------------------------------------------------------
__global__ void __launch_bounds__(256)
k_final(const unsigned short* __restrict__ Mb, const float* __restrict__ x,
        const float* __restrict__ Wb, const float* __restrict__ gamma,
        const float* __restrict__ beta, const float* __restrict__ mean,
        const float* __restrict__ var, float* __restrict__ out) {
  __shared__ unsigned short xt[128 * 130];       // [n][d], padded
  const int tid = threadIdx.x;
  const int bid = blockIdx.x;                    // ((b*8+cht)*512 + nt)
  const int nt = bid & 511, cht = (bid >> 9) & 7, b = bid >> 12;
  const int nbase = nt * 128;
  for (int i = tid; i < 16384; i += 256) {
    int n = i & 127, d = i >> 7;
    xt[n * 130 + d] = f2bf(x[(((size_t)(b * 128 + d)) << 16) + nbase + n]);
  }
  const int wave = tid >> 5, lane = tid & 31;
  const int nn = lane & 15, hl = lane >> 4;
  int kA[8], kB[8];
  a_koffs(hl, kA);
  b_koffs(hl, kB);
  const unsigned short* ma = Mb + ((size_t)(b * 128 + cht * 16 + nn)) * 128;
  __syncthreads();
  v8f acc = {};
#pragma unroll
  for (int kk = 0; kk < 128; kk += 32) {
    v8u A, B;
#pragma unroll
    for (int v = 0; v < 8; ++v) A[v] = *(const unsigned int*)(ma + kk + kA[v]);
    const unsigned short* xb = &xt[(wave * 16 + nn) * 130 + kk];
#pragma unroll
    for (int v = 0; v < 8; ++v) B[v] = *(const unsigned int*)(xb + kB[v]);
    acc = wmma_bf16(A, B, acc);
  }
#pragma unroll
  for (int r = 0; r < 8; ++r) {
    int m = r + hl * 8;
    int c = cht * 16 + m;
    float inv = rsqrtf(var[c] + 1e-5f);
    float s = gamma[c] * inv;
    float sh = beta[c] - mean[c] * s + Wb[c] * s;
    size_t idx = (((size_t)(b * 128 + c)) << 16) + nbase + wave * 16 + nn;
    out[idx] = acc[r] * s + sh + x[idx];
  }
}

// ---------------------------------------------------------------------------
extern "C" void kernel_launch(void* const* d_in, const int* in_sizes, int n_in,
                              void* d_out, int out_size, void* d_ws,
                              size_t ws_size, hipStream_t stream) {
  (void)in_sizes; (void)n_in; (void)out_size; (void)ws_size;
  const float* x     = (const float*)d_in[0];
  const float* thw   = (const float*)d_in[1];
  const float* thb   = (const float*)d_in[2];
  const float* phw   = (const float*)d_in[3];
  const float* phb   = (const float*)d_in[4];
  const float* Ww    = (const float*)d_in[5];
  const float* Wbv   = (const float*)d_in[6];
  const float* gamma = (const float*)d_in[7];
  const float* beta  = (const float*)d_in[8];
  const float* mean  = (const float*)d_in[9];
  const float* var   = (const float*)d_in[10];
  float* out = (float*)d_out;

  char* ws = (char*)d_ws;
  unsigned short* xp  = (unsigned short*)ws; ws += (size_t)8 * 128 * 16384 * 2;
  unsigned short* thB = (unsigned short*)ws; ws += (size_t)8 * 128 * 16384 * 2;
  unsigned short* phB = (unsigned short*)ws; ws += (size_t)8 * 128 * 16384 * 2;
  float*          f   = (float*)ws;          ws += (size_t)8 * 128 * 128 * 4;
  unsigned short* Mb  = (unsigned short*)ws; ws += (size_t)8 * 128 * 128 * 2;
  unsigned short* wrT = (unsigned short*)ws; // theta then phi, 9*128*128 each
  unsigned short* wrP = wrT + (size_t)9 * 128 * 128;

  k_pool   <<<65536, 256, 0, stream>>>(x, xp);
  k_repack <<<1152, 256, 0, stream>>>(thw, phw, wrT);
  k_conv3x3<<<4096, 256, 0, stream>>>(xp, wrT, thb, thB);
  k_conv3x3<<<4096, 256, 0, stream>>>(xp, wrP, phb, phB);
  k_fmm    <<<64, 256, 0, stream>>>(thB, phB, f);
  k_softmax<<<1024, 128, 0, stream>>>(f);
  k_wm     <<<512, 256, 0, stream>>>(Ww, f, Mb);
  k_final  <<<32768, 256, 0, stream>>>(Mb, x, Wbv, gamma, beta, mean, var, out);
}
